// sum_model_75926431859315
// MI455X (gfx1250) — matrix operand order repaired and checked by reading, not compiled, not run on hardware
//
#include <hip/hip_runtime.h>
#include <hip/hip_bf16.h>

typedef __attribute__((ext_vector_type(16))) _Float16 v16h;
typedef __attribute__((ext_vector_type(8)))  _Float16 v8h;
typedef __attribute__((ext_vector_type(8)))  float    v8f;

#define RANK   16
#define HIDDEN 256
#define KSTEPS 6
#define BATCH  256
#define NEMB   64
#define G4H    1024   // 4*HIDDEN

__device__ __forceinline__ float sigf(float x) {
    return 1.0f / (1.0f + __expf(-x));
}
__device__ __forceinline__ float tanhf_(float x) {
    float e = __expf(2.0f * x);
    return (e - 1.0f) / (e + 1.0f);
}
__device__ __forceinline__ v16h cat16(v8h lo, v8h hi) {
    return __builtin_shufflevector(lo, hi, 0, 1, 2, 3, 4, 5, 6, 7,
                                           8, 9, 10, 11, 12, 13, 14, 15);
}

// ---------------------------------------------------------------------------
// WMMA GEMM:  C[M,N] = A[M,K] @ W[N,K]^T + biasScale * bias[N]
// A, W pre-converted to f16 row-major (row stride K halfs = 512B, 16B aligned).
// One wave computes a 16 x (16*NT) output tile; K templated (=256) and fully
// unrolled -> 8*NT v_wmma_f32_16x16x32_f16 back-to-back, A fragment shared
// across the NT B-tiles.
// CDNA5 wave32 layouts:
//   A (16x32 f16): lane l -> m=l&15, q=l>>4; a[0..7]=A[m][kb+q*8+..],
//                  a[8..15]=A[m][kb+16+q*8+..]  (one b128 load each run)
//   B (32x16 f16): lane l -> n=l&15, q=l>>4; b[0..7]=W[n][kb+q*8+..],
//                  b[8..15]=W[n][kb+16+q*8+..]  (two b128)
//   C/D (16x16 f32): acc[r] -> row q*8+r, col l&15
// ---------------------------------------------------------------------------
template <int K, int NT>
__global__ void gemm_awt_wmma(const _Float16* __restrict__ A,
                              const _Float16* __restrict__ W,
                              const float* __restrict__ bias, float biasScale,
                              float* __restrict__ C, int M, int N) {
    int wave = (int)((blockIdx.x * blockDim.x + threadIdx.x) >> 5);
    int lane = (int)(threadIdx.x & 31);
    int groupsN = (N >> 4) / NT;
    int totalGroups = (M >> 4) * groupsN;
    if (wave >= totalGroups) return;          // wave-uniform (EXEC all-1s for WMMA)

    int tm = wave / groupsN;
    int tg = wave - tm * groupsN;
    int q  = lane >> 4;
    int nl = lane & 15;
    int m  = tm * 16 + nl;

    const _Float16* __restrict__ Arow = A + (size_t)m * K;
    const _Float16* __restrict__ Wrow[NT];
#pragma unroll
    for (int t = 0; t < NT; ++t)
        Wrow[t] = W + (size_t)((tg * NT + t) * 16 + nl) * K;

    v8f acc[NT];
#pragma unroll
    for (int t = 0; t < NT; ++t) acc[t] = (v8f){};

#pragma unroll
    for (int kb = 0; kb < K; kb += 32) {
        v8h a0 = *(const v8h*)(Arow + kb + q * 8);
        v8h a1 = *(const v8h*)(Arow + kb + 16 + q * 8);
        v16h a = cat16(a0, a1);
#pragma unroll
        for (int t = 0; t < NT; ++t) {
            v8h b0 = *(const v8h*)(Wrow[t] + kb + q * 8);
            v8h b1 = *(const v8h*)(Wrow[t] + kb + 16 + q * 8);
            v16h b = cat16(b0, b1);
            acc[t] = __builtin_amdgcn_wmma_f32_16x16x32_f16(
                false, a, false, b, (short)0, acc[t], false, false);
        }
    }

#pragma unroll
    for (int t = 0; t < NT; ++t) {
        int n  = (tg * NT + t) * 16 + nl;
        float bv = bias ? biasScale * bias[n] : 0.0f;
#pragma unroll
        for (int r = 0; r < 8; ++r) {
            int row = tm * 16 + q * 8 + r;
            C[(size_t)row * N + n] = acc[t][r] + bv;
        }
    }
}

// f32 -> f16 conversion (weights/embeddings, once per launch)
__global__ void cvt_f16_kernel(const float* __restrict__ src,
                               _Float16* __restrict__ dst, int n) {
    int i = blockIdx.x * blockDim.x + threadIdx.x;
    if (i < n) dst[i] = (_Float16)src[i];
}

// bias2 = bih + bhh
__global__ void add_bias_kernel(const float* __restrict__ bih,
                                const float* __restrict__ bhh,
                                float* __restrict__ bias2) {
    int i = blockIdx.x * blockDim.x + threadIdx.x;
    if (i < G4H) bias2[i] = bih[i] + bhh[i];
}

// Step-0 LSTM cell (c_prev = 0). block=j, thread=u. h emitted in f16 (GEMM A).
__global__ void cell0_kernel(const float* __restrict__ xg,
                             _Float16* __restrict__ h0h,
                             float* __restrict__ c0) {
    int j = blockIdx.x, u = threadIdx.x;
    const float* g = xg + (size_t)j * G4H;
    float i_ = sigf(g[u]);
    float gg = tanhf_(g[2 * HIDDEN + u]);
    float o_ = sigf(g[3 * HIDDEN + u]);
    float c  = i_ * gg;                      // f*0 + i*g
    c0[(size_t)j * HIDDEN + u]  = c;
    h0h[(size_t)j * HIDDEN + u] = (_Float16)(o_ * tanhf_(c));
}

// Normalize out0 (64x16) columns over the 64 rows. Single block, 256 thr.
__global__ void norm0_kernel(float* __restrict__ out0) {
    __shared__ float cs[RANK];
    int t = threadIdx.x;
    if (t < RANK) {
        float s = 0.f;
        for (int j = 0; j < NEMB; ++j) s += out0[j * RANK + t];
        cs[t] = s;
    }
    __syncthreads();
    for (int i = t; i < NEMB * RANK; i += blockDim.x)
        out0[i] /= cs[i & (RANK - 1)];
}

// Gather step-0 state per batch. block=b, thread=u.
__global__ void gather0_kernel(const float* __restrict__ out0,
                               const _Float16* __restrict__ h0h,
                               const float* __restrict__ c0,
                               const int* __restrict__ x_idx,
                               float* __restrict__ total,
                               _Float16* __restrict__ hh,
                               float* __restrict__ c) {
    int b = blockIdx.x, u = threadIdx.x;
    int j = x_idx[b * KSTEPS + 0];
    hh[(size_t)b * HIDDEN + u] = h0h[(size_t)j * HIDDEN + u];
    c[(size_t)b * HIDDEN + u]  = c0[(size_t)j * HIDDEN + u];
    if (u < RANK) total[b * RANK + u] = out0[j * RANK + u];
}

// Cell + j-reduction for steps 1..K-1.  block=b, thread=u.
// hsum[b,u]=sum_j hn (f16, den GEMM A); hsel (f16, num/hg GEMM A); csel f32.
__global__ void cell_mid_kernel(const float* __restrict__ xg,
                                const float* __restrict__ hg,
                                const float* __restrict__ cprev,
                                const int* __restrict__ x_idx, int step,
                                _Float16* __restrict__ hsumh,
                                _Float16* __restrict__ hselh,
                                float* __restrict__ csel) {
    int b = blockIdx.x, u = threadIdx.x;
    int sel = x_idx[b * KSTEPS + step];
    const float* hgb = hg + (size_t)b * G4H;
    float hg0 = hgb[u];
    float hg1 = hgb[HIDDEN + u];
    float hg2 = hgb[2 * HIDDEN + u];
    float hg3 = hgb[3 * HIDDEN + u];
    float cp  = cprev[(size_t)b * HIDDEN + u];

    float hs = 0.f, hv = 0.f, cv = 0.f;
    for (int j = 0; j < NEMB; ++j) {
        const float* xr = xg + (size_t)j * G4H;
        if (j + 1 < NEMB)
            __builtin_prefetch(xr + G4H + u, 0, 0);   // global_prefetch
        float i_ = sigf(xr[u] + hg0);
        float f_ = sigf(xr[HIDDEN + u] + hg1);
        float g_ = tanhf_(xr[2 * HIDDEN + u] + hg2);
        float o_ = sigf(xr[3 * HIDDEN + u] + hg3);
        float cn = f_ * cp + i_ * g_;
        float hn = o_ * tanhf_(cn);
        hs += hn;
        if (j == sel) { hv = hn; cv = cn; }
    }
    hsumh[(size_t)b * HIDDEN + u] = (_Float16)hs;
    hselh[(size_t)b * HIDDEN + u] = (_Float16)hv;
    csel[(size_t)b * HIDDEN + u]  = cv;
}

// total[b] (1x16) = total[b] @ (num[b]/den[b] as 16x16). 1 block, 256 thr.
__global__ void update_total_kernel(const float* __restrict__ num,
                                    const float* __restrict__ den,
                                    float* __restrict__ total) {
    int b = threadIdx.x;
    float t[RANK], nt[RANK];
#pragma unroll
    for (int r = 0; r < RANK; ++r) t[r] = total[b * RANK + r];
#pragma unroll
    for (int s = 0; s < RANK; ++s) {
        float acc = 0.f;
        for (int r = 0; r < RANK; ++r) {
            int o = b * RANK * RANK + r * RANK + s;
            acc += t[r] * (num[o] / den[o]);
        }
        nt[s] = acc;
    }
#pragma unroll
    for (int s = 0; s < RANK; ++s) total[b * RANK + s] = nt[s];
}

// out[b] = sum_r total[b,r] * numL[b,r]/denL[b,r].  1 block, 256 thr.
__global__ void finalize_kernel(const float* __restrict__ numL,
                                const float* __restrict__ denL,
                                const float* __restrict__ total,
                                float* __restrict__ out) {
    int b = threadIdx.x;
    float acc = 0.f;
#pragma unroll
    for (int r = 0; r < RANK; ++r)
        acc += total[b * RANK + r] * (numL[b * RANK + r] / denL[b * RANK + r]);
    out[b] = acc;
}

extern "C" void kernel_launch(void* const* d_in, const int* in_sizes, int n_in,
                              void* d_out, int out_size, void* d_ws, size_t ws_size,
                              hipStream_t stream) {
    const int*   x_idx = (const int*)d_in[0];
    const float* emb   = (const float*)d_in[1];
    const float* Wih   = (const float*)d_in[2];
    const float* Whh   = (const float*)d_in[3];
    const float* bih   = (const float*)d_in[4];
    const float* bhh   = (const float*)d_in[5];
    const float* Wf    = (const float*)d_in[6];
    const float* bf    = (const float*)d_in[7];
    const float* Wm    = (const float*)d_in[8];
    const float* bm    = (const float*)d_in[9];
    const float* Wl    = (const float*)d_in[10];
    const float* bl    = (const float*)d_in[11];
    float* out = (float*)d_out;

    // ---------------- workspace carve-up ----------------
    float* ws    = (float*)d_ws;
    float* bias2 = ws;                   // 1024
    float* xg    = bias2 + 1024;         // 64*1024
    float* c0    = xg + NEMB * G4H;      // 64*256
    float* out0  = c0 + NEMB * HIDDEN;   // 64*16
    float* cA    = out0 + NEMB * RANK;   // 256*256
    float* cB    = cA + BATCH * HIDDEN;  // 256*256
    float* hg    = cB + BATCH * HIDDEN;  // 256*1024
    float* numb  = hg + BATCH * G4H;     // 256*256
    float* den   = numb + BATCH * HIDDEN;// 256*256
    float* total = den + BATCH * HIDDEN; // 256*16
    float* numL  = total + BATCH * RANK; // 256*16
    float* denL  = numL + BATCH * RANK;  // 256*16
    float* f32end = denL + BATCH * RANK;

    _Float16* hws   = (_Float16*)f32end;    // 16B-aligned (all sizes mult. of 8)
    _Float16* embh  = hws;                  // 64*256
    _Float16* Wihh  = embh + NEMB * HIDDEN; // 1024*256
    _Float16* Whhh  = Wihh + G4H * HIDDEN;  // 1024*256
    _Float16* Wmh   = Whhh + G4H * HIDDEN;  // 256*256
    _Float16* Wfh   = Wmh + HIDDEN * HIDDEN;// 16*256
    _Float16* Wlh   = Wfh + RANK * HIDDEN;  // 16*256
    _Float16* h0h   = Wlh + RANK * HIDDEN;  // 64*256
    _Float16* hAh   = h0h + NEMB * HIDDEN;  // 256*256
    _Float16* hBh   = hAh + BATCH * HIDDEN; // 256*256
    _Float16* hsumh = hBh + BATCH * HIDDEN; // 256*256

    // ---------------- one-time conversions ----------------
    cvt_f16_kernel<<<(NEMB * HIDDEN + 255) / 256, 256, 0, stream>>>(emb, embh, NEMB * HIDDEN);
    cvt_f16_kernel<<<(G4H * HIDDEN + 255) / 256, 256, 0, stream>>>(Wih, Wihh, G4H * HIDDEN);
    cvt_f16_kernel<<<(G4H * HIDDEN + 255) / 256, 256, 0, stream>>>(Whh, Whhh, G4H * HIDDEN);
    cvt_f16_kernel<<<(HIDDEN * HIDDEN + 255) / 256, 256, 0, stream>>>(Wm, Wmh, HIDDEN * HIDDEN);
    cvt_f16_kernel<<<(RANK * HIDDEN + 255) / 256, 256, 0, stream>>>(Wf, Wfh, RANK * HIDDEN);
    cvt_f16_kernel<<<(RANK * HIDDEN + 255) / 256, 256, 0, stream>>>(Wl, Wlh, RANK * HIDDEN);
    add_bias_kernel<<<4, 256, 0, stream>>>(bih, bhh, bias2);

    // xg = emb @ Wih^T + bias2  (M=64,N=1024,NT=2 -> 128 tile-groups)
    gemm_awt_wmma<HIDDEN, 2><<<16, 256, 0, stream>>>(embh, Wihh, bias2, 1.f, xg, NEMB, G4H);

    // step 0
    cell0_kernel<<<NEMB, HIDDEN, 0, stream>>>(xg, h0h, c0);
    gemm_awt_wmma<HIDDEN, 1><<<1, 256, 0, stream>>>(h0h, Wfh, bf, 1.f, out0, NEMB, RANK);
    norm0_kernel<<<1, 256, 0, stream>>>(out0);
    gather0_kernel<<<BATCH, HIDDEN, 0, stream>>>(out0, h0h, c0, x_idx, total, hAh, cA);

    _Float16 *hcur = hAh, *hnext = hBh;
    float    *ccur = cA,  *cnext = cB;

    for (int step = 1; step <= KSTEPS - 2; ++step) {
        // hg = h @ Whh^T  (M=256,N=1024,NT=2 -> 512 groups)
        gemm_awt_wmma<HIDDEN, 2><<<64, 256, 0, stream>>>(hcur, Whhh, nullptr, 0.f,
                                                         hg, BATCH, G4H);
        cell_mid_kernel<<<BATCH, HIDDEN, 0, stream>>>(xg, hg, ccur, x_idx, step,
                                                      hsumh, hnext, cnext);
        // num = hsel@Wm^T + bm ; den = hsum@Wm^T + 64*bm  (M=256,N=256,NT=2 -> 128)
        gemm_awt_wmma<HIDDEN, 2><<<16, 256, 0, stream>>>(hnext, Wmh, bm, 1.f,
                                                         numb, BATCH, HIDDEN);
        gemm_awt_wmma<HIDDEN, 2><<<16, 256, 0, stream>>>(hsumh, Wmh, bm, 64.f,
                                                         den, BATCH, HIDDEN);
        update_total_kernel<<<1, 256, 0, stream>>>(numb, den, total);
        _Float16* th = hcur; hcur = hnext; hnext = th;
        float*    tc = ccur; ccur = cnext; cnext = tc;
    }

    // last step
    gemm_awt_wmma<HIDDEN, 2><<<64, 256, 0, stream>>>(hcur, Whhh, nullptr, 0.f,
                                                     hg, BATCH, G4H);
    cell_mid_kernel<<<BATCH, HIDDEN, 0, stream>>>(xg, hg, ccur, x_idx, KSTEPS - 1,
                                                  hsumh, hnext, cnext);
    gemm_awt_wmma<HIDDEN, 1><<<2, 256, 0, stream>>>(hnext, Wlh, bl, 1.f,
                                                    numL, BATCH, RANK);
    gemm_awt_wmma<HIDDEN, 1><<<2, 256, 0, stream>>>(hsumh, Wlh, bl, 64.f,
                                                    denL, BATCH, RANK);
    finalize_kernel<<<1, 256, 0, stream>>>(numL, denL, total, out);
}
